// iresgnn_5394478923811
// MI455X (gfx1250) — matrix operand order repaired and checked by reading, not compiled, not run on hardware
//
#include <hip/hip_runtime.h>
#include <cstdint>

#define N_NODES 40000
#define N_EDGES 640000
#define HD      128
#define OUTD    10
#define THRV    1e-4f
#define LN_EPS  1e-5f

typedef __attribute__((ext_vector_type(16))) __bf16 v16bf;
typedef __attribute__((ext_vector_type(8)))  __bf16 v8bf;
typedef __attribute__((ext_vector_type(8)))  float  v8f;

#if __has_builtin(__builtin_amdgcn_global_load_async_to_lds_b128)
#define HAVE_ASYNC_LDS 1
// exact parameter types per hipcc diagnostic: (int4* AS1, int4* AS3, imm, imm)
typedef int v4i_n __attribute__((vector_size(16)));
typedef __attribute__((address_space(1))) v4i_n* as1_v4i;
typedef __attribute__((address_space(3))) v4i_n* as3_v4i;
#else
#define HAVE_ASYNC_LDS 0
#endif

__device__ __forceinline__ void wait_asynccnt0() {
#if __has_builtin(__builtin_amdgcn_s_wait_asynccnt)
  __builtin_amdgcn_s_wait_asynccnt(0);
#else
  asm volatile("s_wait_asynccnt 0x0" ::: "memory");
#endif
}

__device__ __forceinline__ unsigned short f2bf_bits(float f) {
  unsigned u = __float_as_uint(f);
  unsigned r = u + 0x7FFFu + ((u >> 16) & 1u);   // round-to-nearest-even
  return (unsigned short)(r >> 16);
}

// ---------------- setup kernels ----------------

__global__ void k_init_flags(float* diff, int* done, int* zero) {
  *diff = 0.f; *done = 0; *zero = 0;
}

__global__ void k_edges32(const long long* __restrict__ eidx,
                          int* __restrict__ srcs, int* __restrict__ dsts) {
  int i = blockIdx.x * blockDim.x + threadIdx.x;
  if (i >= N_EDGES) return;
  srcs[i] = (int)eidx[i];
  dsts[i] = (int)eidx[N_EDGES + i];
}

__global__ void k_fill1(float* __restrict__ p, int n) {
  int i = blockIdx.x * blockDim.x + threadIdx.x;
  if (i < n) p[i] = 1.0f;                     // self-loop contribution to degree
}

__global__ void k_degcount(const int* __restrict__ dsts, float* __restrict__ deg) {
  int i = blockIdx.x * blockDim.x + threadIdx.x;
  if (i >= N_EDGES) return;
  unsafeAtomicAdd(&deg[dsts[i]], 1.0f);       // hardware global_atomic_add_f32
}

__global__ void k_rsqrt(float* __restrict__ p, int n) {
  int i = blockIdx.x * blockDim.x + threadIdx.x;
  if (i < n) p[i] = rsqrtf(p[i]);
}

__global__ void k_norm(const int* __restrict__ srcs, const int* __restrict__ dsts,
                       const float* __restrict__ dinv, float* __restrict__ norm, int total) {
  int i = blockIdx.x * blockDim.x + threadIdx.x;
  if (i >= total) return;
  int a, b;
  if (i < N_EDGES) { a = srcs[i]; b = dsts[i]; } else { a = b = i - N_EDGES; }
  norm[i] = dinv[a] * dinv[b];
}

// Wt[n*K + k] = bf16(W[k*Nc + n]); Nc fixed 128.
__global__ void k_tr_bf16(const float* __restrict__ W, unsigned short* __restrict__ Wt, int K) {
  int tid = blockIdx.x * blockDim.x + threadIdx.x;
  if (tid >= K * HD) return;
  int n = tid / K, k = tid - n * K;
  Wt[tid] = f2bf_bits(W[k * HD + n]);
}

// f32 -> bf16, 4 elements / thread, gated on done flag
__global__ void k_cvt(const float* __restrict__ in, unsigned short* __restrict__ out,
                      int n4, const int* __restrict__ done) {
  if (*done) return;
  int i = blockIdx.x * blockDim.x + threadIdx.x;
  if (i >= n4) return;
  float4 v = ((const float4*)in)[i];
  uint2 r;
  r.x = (unsigned)f2bf_bits(v.x) | ((unsigned)f2bf_bits(v.y) << 16);
  r.y = (unsigned)f2bf_bits(v.z) | ((unsigned)f2bf_bits(v.w) << 16);
  ((uint2*)out)[i] = r;
}

__global__ void k_copy4(float* __restrict__ dst, const float* __restrict__ src, int n4) {
  int i = blockIdx.x * blockDim.x + threadIdx.x;
  if (i < n4) ((float4*)dst)[i] = ((const float4*)src)[i];
}

// ---------------- WMMA GEMM: C[M,128] = act(A_bf16[M,K] @ Wt_bf16^T + bias) ----------------
// Wt layout: [n][k] (pre-transposed). mode 0: raw, mode 1: bias + relu.
__global__ __launch_bounds__(128) void k_gemm(const __bf16* __restrict__ A,
                                              const __bf16* __restrict__ Wt,
                                              const float* __restrict__ bias,
                                              float* __restrict__ C,
                                              int K, int mode,
                                              const int* __restrict__ done) {
  if (*done) return;
  __shared__ __align__(32) __bf16 sW[HD * 128];   // 32 KB chunk: [n 0..127][kk 0..127]
  const int lane = threadIdx.x & 31;
  const int wave = threadIdx.x >> 5;
  const int rowBase = blockIdx.x * 64 + wave * 16;
  const int n    = lane & 15;
  const int half = lane >> 4;

  v8f acc[8] = {};

  for (int k0 = 0; k0 < K; k0 += 128) {
    __syncthreads();
    // stage Wt[:, k0:k0+128] -> LDS, 16 bytes per lane per step
#if HAVE_ASYNC_LDS
    for (int idx = threadIdx.x; idx < HD * 16; idx += blockDim.x) {
      int nrow  = idx >> 4;
      int chunk = idx & 15;
      as1_v4i gsrc = (as1_v4i)(Wt + (size_t)nrow * K + k0 + chunk * 8);
      as3_v4i ldst = (as3_v4i)(sW + (size_t)idx * 8);
      __builtin_amdgcn_global_load_async_to_lds_b128(gsrc, ldst, 0, 0);
    }
    wait_asynccnt0();
#else
    for (int idx = threadIdx.x; idx < HD * 16; idx += blockDim.x) {
      int nrow  = idx >> 4;
      int chunk = idx & 15;
      *(uint4*)(sW + (size_t)idx * 8) =
          *(const uint4*)(Wt + (size_t)nrow * K + k0 + chunk * 8);
    }
#endif
    __syncthreads();

    const int mrow = rowBase + (lane & 15);
    #pragma unroll
    for (int kb = 0; kb < 4; ++kb) {
      // A fragment: 16x32 bf16, element s holds k = 16*half + s (same k order as B)
      const __bf16* arow = A + (size_t)mrow * K + k0 + kb * 32;
      v8bf lo = *(const v8bf*)(arow + 16 * half);
      v8bf hi = *(const v8bf*)(arow + 16 * half + 8);
      v16bf afrag;
      #pragma unroll
      for (int j = 0; j < 8; ++j) { afrag[j] = lo[j]; afrag[8 + j] = hi[j]; }

      // load all 8 B fragments into distinct registers so ds_loads clause/pipeline
      v16bf bfr[8];
      #pragma unroll
      for (int ct = 0; ct < 8; ++ct)
        bfr[ct] = *(const v16bf*)(sW + (size_t)(ct * 16 + n) * 128 + kb * 32 + 16 * half);

      #pragma unroll
      for (int ct = 0; ct < 8; ++ct)
        acc[ct] = __builtin_amdgcn_wmma_f32_16x16x32_bf16(
            false, afrag, false, bfr[ct], (short)0, acc[ct], false, false);
    }
  }

  // writeback (C layout per ISA: element r -> row rowBase + r + 8*half, col ct*16+n)
  #pragma unroll
  for (int ct = 0; ct < 8; ++ct) {
    int col = ct * 16 + n;
    float bc = bias[col];
    #pragma unroll
    for (int r = 0; r < 8; ++r) {
      int m = rowBase + r + 8 * half;
      float v = acc[ct][r];
      if (mode == 1) v = fmaxf(v + bc, 0.f);
      C[(size_t)m * HD + col] = v;
    }
  }
}

// ---------------- GCN aggregate ----------------

__global__ void k_biasfill(float* __restrict__ t2, const float* __restrict__ b,
                           const int* __restrict__ done) {
  if (*done) return;
  int i = blockIdx.x * blockDim.x + threadIdx.x;
  if (i < N_NODES * HD) t2[i] = b[i & (HD - 1)];
}

// one wave per message; float4 gather + 4 hw fp32 atomics per lane
__global__ void k_scatter(const float* __restrict__ hsrc, const float* __restrict__ norm,
                          const int* __restrict__ srcs, const int* __restrict__ dsts,
                          float* __restrict__ out, const int* __restrict__ done) {
  if (*done) return;
  int msg  = blockIdx.x * (blockDim.x >> 5) + (threadIdx.x >> 5);
  int lane = threadIdx.x & 31;
  if (msg >= N_EDGES + N_NODES) return;
  int s, d;
  if (msg < N_EDGES) { s = srcs[msg]; d = dsts[msg]; } else { s = d = msg - N_EDGES; }
  float w = norm[msg];
  float4 v = *(const float4*)(hsrc + (size_t)s * HD + lane * 4);
  float* o = out + (size_t)d * HD + lane * 4;
  unsafeAtomicAdd(o + 0, w * v.x);
  unsafeAtomicAdd(o + 1, w * v.y);
  unsafeAtomicAdd(o + 2, w * v.z);
  unsafeAtomicAdd(o + 3, w * v.w);
}

__global__ void k_addrelu(float* __restrict__ zf, const float* __restrict__ t2) {
  int i = blockIdx.x * blockDim.x + threadIdx.x;
  if (i < N_NODES * HD) zf[i] += fmaxf(t2[i], 0.f);
}

// inverse step: x <- x_input - relu(agg); accumulate sum|delta|
__global__ void k_update(const float* __restrict__ xin, const float* __restrict__ t2,
                         float* __restrict__ xc, float* __restrict__ diff,
                         const int* __restrict__ done) {
  if (*done) return;
  int i = blockIdx.x * blockDim.x + threadIdx.x;
  float d = 0.f;
  if (i < N_NODES * HD) {
    float xn = xin[i] - fmaxf(t2[i], 0.f);
    d = fabsf(xn - xc[i]);
    xc[i] = xn;
  }
  #pragma unroll
  for (int off = 16; off; off >>= 1) d += __shfl_xor(d, off, 32);
  if ((threadIdx.x & 31) == 0) unsafeAtomicAdd(diff, d);
}

__global__ void k_check(float* diff, int* done) {
  if (*diff < THRV * (float)(N_NODES * HD)) *done = 1;
  *diff = 0.f;
}

__global__ void k_reset(float* diff, int* done) { *diff = 0.f; *done = 0; }

// ---------------- LayerNorm: concat(zf,zb) -> normalized bf16 [N,256] ----------------
__global__ void k_ln(const float* __restrict__ zf, const float* __restrict__ zb,
                     const float* __restrict__ g, const float* __restrict__ be,
                     unsigned short* __restrict__ out) {
  int row  = blockIdx.x * (blockDim.x >> 5) + (threadIdx.x >> 5);
  int lane = threadIdx.x & 31;
  if (row >= N_NODES) return;
  float v[8], s = 0.f, ss = 0.f;
  #pragma unroll
  for (int j = 0; j < 8; ++j) {
    int col = lane * 8 + j;
    float x = (col < HD) ? zf[(size_t)row * HD + col] : zb[(size_t)row * HD + col - HD];
    v[j] = x; s += x; ss += x * x;
  }
  #pragma unroll
  for (int off = 16; off; off >>= 1) { s += __shfl_xor(s, off, 32); ss += __shfl_xor(ss, off, 32); }
  float mu  = s * (1.f / 256.f);
  float var = ss * (1.f / 256.f) - mu * mu;
  float inv = rsqrtf(var + LN_EPS);
  #pragma unroll
  for (int j = 0; j < 8; ++j) {
    int col = lane * 8 + j;
    float y = (v[j] - mu) * inv * g[col] + be[col];
    out[(size_t)row * 256 + col] = f2bf_bits(y);
  }
}

// ---------------- final tiny projection: out = out1 @ W2 + b2 ----------------
__global__ void k_out(const float* __restrict__ out1, const float* __restrict__ W2,
                      const float* __restrict__ b2, float* __restrict__ out) {
  __shared__ float sw[HD * OUTD];
  for (int i = threadIdx.x; i < HD * OUTD; i += blockDim.x) sw[i] = W2[i];
  __syncthreads();
  int node = blockIdx.x * blockDim.x + threadIdx.x;
  if (node >= N_NODES) return;
  float acc[OUTD];
  #pragma unroll
  for (int o = 0; o < OUTD; ++o) acc[o] = b2[o];
  const float* row = out1 + (size_t)node * HD;
  for (int k = 0; k < HD; ++k) {
    float xv = row[k];
    #pragma unroll
    for (int o = 0; o < OUTD; ++o) acc[o] += xv * sw[k * OUTD + o];
  }
  #pragma unroll
  for (int o = 0; o < OUTD; ++o) out[(size_t)node * OUTD + o] = acc[o];
}

// ---------------- host launch ----------------

extern "C" void kernel_launch(void* const* d_in, const int* in_sizes, int n_in,
                              void* d_out, int out_size, void* d_ws, size_t ws_size,
                              hipStream_t stream) {
  const float*     x      = (const float*)d_in[0];
  const long long* eidx   = (const long long*)d_in[1];
  const float*     W_in   = (const float*)d_in[2];
  const float*     b_in   = (const float*)d_in[3];
  const float*     W0     = (const float*)d_in[4];
  const float*     b0     = (const float*)d_in[5];
  const float*     W1     = (const float*)d_in[6];
  const float*     b1     = (const float*)d_in[7];
  const float*     ln_g   = (const float*)d_in[8];
  const float*     ln_b   = (const float*)d_in[9];
  const float*     W_out1 = (const float*)d_in[10];
  const float*     b_out1 = (const float*)d_in[11];
  const float*     W_out2 = (const float*)d_in[12];
  const float*     b_out2 = (const float*)d_in[13];
  float* out = (float*)d_out;

  char* ws = (char*)d_ws;
  size_t off = 0;
  auto alloc = [&](size_t bytes) -> char* {
    off = (off + 255) & ~(size_t)255;
    char* p = ws + off; off += bytes; return p;
  };

  const size_t NH = (size_t)N_NODES * HD;
  float* dinv  = (float*)alloc((size_t)N_NODES * 4);
  int*   srcs  = (int*)alloc((size_t)N_EDGES * 4);
  int*   dsts  = (int*)alloc((size_t)N_EDGES * 4);
  float* norm  = (float*)alloc((size_t)(N_EDGES + N_NODES) * 4);
  unsigned short* act16 = (unsigned short*)alloc((size_t)N_NODES * 256 * 2);
  float* hbuf  = (float*)alloc(NH * 4);   // reused as xc after zf/zb init
  float* zf    = (float*)alloc(NH * 4);
  float* zb    = (float*)alloc(NH * 4);
  float* tmp1  = (float*)alloc(NH * 4);
  float* tmp2  = (float*)alloc(NH * 4);
  unsigned short* WtIn = (unsigned short*)alloc((size_t)HD * HD * 2);
  unsigned short* Wt0  = (unsigned short*)alloc((size_t)HD * HD * 2);
  unsigned short* Wt1  = (unsigned short*)alloc((size_t)HD * HD * 2);
  unsigned short* WtO1 = (unsigned short*)alloc((size_t)256 * HD * 2);
  float* diff  = (float*)alloc(4);
  int*   done  = (int*)alloc(4);
  int*   zero  = (int*)alloc(4);
  float* xc    = hbuf;

  const int T = 256;
  const int gE    = N_EDGES / T;                    // 2500
  const int gN    = (N_NODES + T - 1) / T;          // 157
  const int gMsg  = (N_EDGES + N_NODES) / 8;        // 85000 (wave per message)
  const int gNrm  = (N_EDGES + N_NODES + T - 1) / T;
  const int gCvt  = (int)(NH / 4 / T);              // 5000
  const int gEl   = (int)(NH / T);                  // 20000
  const int gGemm = N_NODES / 64;                   // 625
  const int gLn   = N_NODES / 8;                    // 5000

  // ---- graph setup ----
  k_init_flags<<<1, 1, 0, stream>>>(diff, done, zero);
  k_edges32<<<gE, T, 0, stream>>>(eidx, srcs, dsts);
  k_fill1<<<gN, T, 0, stream>>>(dinv, N_NODES);
  k_degcount<<<gE, T, 0, stream>>>(dsts, dinv);
  k_rsqrt<<<gN, T, 0, stream>>>(dinv, N_NODES);
  k_norm<<<gNrm, T, 0, stream>>>(srcs, dsts, dinv, norm, N_EDGES + N_NODES);

  // ---- weights -> transposed bf16 ----
  k_tr_bf16<<<(HD * HD + T - 1) / T, T, 0, stream>>>(W_in, WtIn, HD);
  k_tr_bf16<<<(HD * HD + T - 1) / T, T, 0, stream>>>(W0, Wt0, HD);
  k_tr_bf16<<<(HD * HD + T - 1) / T, T, 0, stream>>>(W1, Wt1, HD);
  k_tr_bf16<<<(256 * HD + T - 1) / T, T, 0, stream>>>(W_out1, WtO1, 256);

  // ---- h = relu(x @ W_in + b_in) ----
  k_cvt<<<gCvt, T, 0, stream>>>(x, act16, (int)(NH / 4), zero);
  k_gemm<<<gGemm, 128, 0, stream>>>((const __bf16*)act16, (const __bf16*)WtIn, b_in,
                                    hbuf, HD, 1, zero);
  k_copy4<<<gCvt, T, 0, stream>>>(zf, hbuf, (int)(NH / 4));
  k_copy4<<<gCvt, T, 0, stream>>>(zb, hbuf, (int)(NH / 4));

  // ---- forward residual: 4 GCN passes ----
  for (int d = 0; d < 4; ++d) {
    const unsigned short* Wt = (d % 2 == 0) ? Wt0 : Wt1;
    const float* bb = (d % 2 == 0) ? b0 : b1;
    k_cvt<<<gCvt, T, 0, stream>>>(zf, act16, (int)(NH / 4), zero);
    k_gemm<<<gGemm, 128, 0, stream>>>((const __bf16*)act16, (const __bf16*)Wt, bb,
                                      tmp1, HD, 0, zero);
    k_biasfill<<<gEl, T, 0, stream>>>(tmp2, bb, zero);
    k_scatter<<<gMsg, T, 0, stream>>>(tmp1, norm, srcs, dsts, tmp2, zero);
    k_addrelu<<<gEl, T, 0, stream>>>(zf, tmp2);
  }

  // ---- inverse blocks: 4 x (up to 8 gated fixed-point iterations) ----
  for (int i = 0; i < 4; ++i) {
    const unsigned short* Wt = (i % 2 == 0) ? Wt1 : Wt0;   // blocks[1 - i%2]
    const float* bb = (i % 2 == 0) ? b1 : b0;
    k_copy4<<<gCvt, T, 0, stream>>>(xc, zb, (int)(NH / 4));
    k_reset<<<1, 1, 0, stream>>>(diff, done);
    for (int it = 0; it < 8; ++it) {
      k_cvt<<<gCvt, T, 0, stream>>>(xc, act16, (int)(NH / 4), done);
      k_gemm<<<gGemm, 128, 0, stream>>>((const __bf16*)act16, (const __bf16*)Wt, bb,
                                        tmp1, HD, 0, done);
      k_biasfill<<<gEl, T, 0, stream>>>(tmp2, bb, done);
      k_scatter<<<gMsg, T, 0, stream>>>(tmp1, norm, srcs, dsts, tmp2, done);
      k_update<<<gEl, T, 0, stream>>>(zb, tmp2, xc, diff, done);
      k_check<<<1, 1, 0, stream>>>(diff, done);
    }
    k_copy4<<<gCvt, T, 0, stream>>>(zb, xc, (int)(NH / 4));
  }

  // ---- LayerNorm(concat) -> bf16 [N,256], head GEMM, final projection ----
  k_ln<<<gLn, T, 0, stream>>>(zf, zb, ln_g, ln_b, act16);
  k_gemm<<<gGemm, 128, 0, stream>>>((const __bf16*)act16, (const __bf16*)WtO1, b_out1,
                                    tmp1, 256, 1, zero);
  k_out<<<gN, T, 0, stream>>>(tmp1, W_out2, b_out2, out);
}
